// MultiHeadAttention_14731737825340
// MI455X (gfx1250) — matrix operand order repaired and checked by reading, not compiled, not run on hardware
//
#include <hip/hip_runtime.h>

typedef __bf16 bf16;
typedef __attribute__((ext_vector_type(4)))  __bf16 v4bf;
typedef __attribute__((ext_vector_type(8)))  __bf16 v8bf;
typedef __attribute__((ext_vector_type(16))) __bf16 v16bf;
typedef __attribute__((ext_vector_type(4)))  float  v4f;
typedef __attribute__((ext_vector_type(8)))  float  v8f;
typedef __attribute__((ext_vector_type(4)))  unsigned int v4u;
typedef __attribute__((ext_vector_type(4)))  int v4i;
typedef __attribute__((ext_vector_type(8)))  int v8i;

#define D_MODEL 1024
#define N_HEAD  16
#define D_HEAD  64
#define BATCH   2
#define SEQ     2048
#define M_TOT   (BATCH*SEQ)          // 4096 total rows

#define BPAD 40   // 32 + 8 bf16 pad: 80B row stride (16 DW + 4 DW), conflict-free
#define KPAD 72   // 64 + 8 bf16 pad: 144B row stride (32 DW + 4 DW)

// ---------------- scheduler pipeline hints (best-effort) --------------------
#if __has_builtin(__builtin_amdgcn_sched_group_barrier)
#define SCHED_GROUP(mask, n, id) __builtin_amdgcn_sched_group_barrier((mask), (n), (id))
#else
#define SCHED_GROUP(mask, n, id)
#endif
// masks: 0x020 VMEM read, 0x100 DS read, 0x200 DS write, 0x008 matrix (WMMA)

// ------------------- Tensor Data Mover (TDM) support ------------------------
#if __has_builtin(__builtin_amdgcn_tensor_load_to_lds)
#define USE_TDM 1
#if defined(__clang_major__) && __clang_major__ >= 23
#define TDM_CALL(g0, g1, g2, g3) \
  __builtin_amdgcn_tensor_load_to_lds((g0), (g1), (g2), (g3), (v8i){0,0,0,0,0,0,0,0}, 0)
#else
#define TDM_CALL(g0, g1, g2, g3) \
  __builtin_amdgcn_tensor_load_to_lds((g0), (g1), (g2), (g3), 0)
#endif

// D# group1 word0: data_size=2B, pad_enable, pad_interval, pad_amount=4 DW
#define TDM_FLAGS_PAD16 ((1 << 16) | (1 << 20) | (3 << 22) | (3 << 25)) // 16-DW rows
#define TDM_FLAGS_PAD32 ((1 << 16) | (1 << 20) | (4 << 22) | (3 << 25)) // 32-DW rows

// 2D tile DMA: global (row stride stride0 elems) -> LDS (padded rows).
// dims/tiles in elements; dims sized == tile so no OOB path.
static __device__ __forceinline__ void tdm_load_2d(
    unsigned lds_addr, unsigned long long gaddr, int flags,
    int dim0, int dim1, int tile0, int tile1, int stride0) {
  v4u g0;
  g0[0] = 1u;                                    // count=1, user descriptor
  g0[1] = lds_addr;                              // LDS byte address
  g0[2] = (unsigned)gaddr;                       // global addr [31:0]
  g0[3] = ((unsigned)(gaddr >> 32) & 0x01FFFFFFu) | 0x80000000u; // [56:32] | type=2
  v8i g1;
  g1[0] = flags;
  g1[1] = (int)(((unsigned)dim0 & 0xFFFFu) << 16);                       // dim0 lo16
  g1[2] = (int)(((unsigned)dim0 >> 16) | (((unsigned)dim1 & 0xFFFFu) << 16));
  g1[3] = (int)(((unsigned)dim1 >> 16) | ((unsigned)tile0 << 16));       // tile_dim0
  g1[4] = tile1;                                 // tile_dim1 (tile_dim2 = 0)
  g1[5] = stride0;                               // tensor_dim0_stride lo32
  g1[6] = 0;
  g1[7] = 0;
  v4i gz; gz[0] = 0; gz[1] = 0; gz[2] = 0; gz[3] = 0;
  TDM_CALL(g0, g1, gz, gz);
}
#endif

static __device__ __forceinline__ v16bf cat16(v8bf lo, v8bf hi) {
  v16bf r;
#pragma unroll
  for (int i = 0; i < 8; ++i) { r[i] = lo[i]; r[i + 8] = hi[i]; }
  return r;
}

// ---------------- elementwise f32 -> bf16 (row-major preserved) -------------
__global__ void k_cvt_bf16(const float* __restrict__ src, bf16* __restrict__ dst, int n) {
  int i = (blockIdx.x * blockDim.x + threadIdx.x) * 4;
  if (i + 3 < n) {
    v4f v = *(const v4f*)(src + i);
    v4bf o;
    o[0] = (bf16)v[0]; o[1] = (bf16)v[1]; o[2] = (bf16)v[2]; o[3] = (bf16)v[3];
    *(v4bf*)(dst + i) = o;
  }
}

// ------------- weight transpose+convert: Wt[n][k] = (bf16)W[k][n] -----------
__global__ void k_wT_bf16(const float* __restrict__ w, bf16* __restrict__ wt) {
  int idx = blockIdx.x * blockDim.x + threadIdx.x;     // 0 .. 1M-1
  int k = idx & (D_MODEL - 1);
  int n = idx >> 10;
  wt[idx] = (bf16)w[k * D_MODEL + n];
}

// ------------------------- bf16 WMMA GEMM -----------------------------------
// C[M,N] = (A[M,K] * Bt[N,K]^T + bias) * oscale.  M=4096, N=K=1024.
// Block = 8 waves -> 128 rows x 64 cols. B tile (64n x 32k) staged in
// double-buffered LDS via TDM (wave 0 issues DMA), consumed by all 8 waves.
// A operand register-pipelined one k-step ahead.
// mode 0: out_bf = per-head [B,H,L,64]            (Q,K projections)
// mode 1: out_bf = per-head transposed [B,H,64,L] (V projection)
// mode 2: out_f  = row-major f32 [M,N]            (final output projection)
__global__ __launch_bounds__(256, 2) void k_gemm_bf16(
    const bf16* __restrict__ A, const bf16* __restrict__ Bt,
    const float* __restrict__ bias,
    bf16* __restrict__ out_bf, float* __restrict__ out_f,
    int mode, float oscale) {
  __shared__ bf16 bsh[2][64][BPAD];    // [buf][n][k]

  const int tid  = threadIdx.x;
  const int w    = tid >> 5;
  const int lane = tid & 31;
  const int half = lane >> 4;
  const int l15  = lane & 15;
  const int m0 = blockIdx.x * 128 + w * 16;
  const int n0 = blockIdx.y * 64;

  const bf16* aptr = A + (size_t)(m0 + l15) * D_MODEL;
  v8f acc[4] = {};

#ifdef USE_TDM
  const unsigned ldsB0 = (unsigned)(size_t)&bsh[0][0][0];
  const unsigned ldsB1 = (unsigned)(size_t)&bsh[1][0][0];
  const unsigned long long gB =
      (unsigned long long)(size_t)(Bt + (size_t)n0 * D_MODEL);
  if (w == 0) {
    tdm_load_2d(ldsB0, gB, TDM_FLAGS_PAD16, 32, 64, 32, 64, D_MODEL);
    __builtin_amdgcn_s_wait_tensorcnt(0);
  }
  __syncthreads();
#else
  const int fn = tid >> 2;            // 0..63
  const int fk = (tid & 3) * 8;       // 0,8,16,24
  const bf16* bsrc = Bt + (size_t)(n0 + fn) * D_MODEL + fk;
  *(v8bf*)(&bsh[0][fn][fk]) = *(const v8bf*)(bsrc);
  __syncthreads();
#endif

  // A operand, 16x32 layout: lanes 0-15 K{0..7,16..23}, lanes 16-31 K{8..15,24..31}
  v16bf aop = cat16(*(const v8bf*)(aptr + 8 * half),
                    *(const v8bf*)(aptr + 16 + 8 * half));

  int buf = 0;
  for (int kk = 0; kk < D_MODEL; kk += 32, buf ^= 1) {
#ifdef USE_TDM
    if (w == 0 && kk + 32 < D_MODEL)
      tdm_load_2d(buf ? ldsB0 : ldsB1, gB + (unsigned)((kk + 32) * 2),
                  TDM_FLAGS_PAD16, 32, 64, 32, 64, D_MODEL);
#else
    const int kkb = (kk + 32 < D_MODEL) ? kk + 32 : 0;
    v8bf bnext = *(const v8bf*)(bsrc + kkb);
#endif
    __builtin_prefetch(aptr + kk + 64, 0, 3);

    // next A operand issued a full iteration ahead (clamped on last iter)
    const int kkn = (kk + 32 < D_MODEL) ? kk + 32 : 0;
    v16bf aop_n = cat16(*(const v8bf*)(aptr + kkn + 8 * half),
                        *(const v8bf*)(aptr + kkn + 16 + 8 * half));

    // hoist all B operands so the ds loads clause together
    v16bf bop[4];
#pragma unroll
    for (int c = 0; c < 4; ++c) {
      const bf16* bp = &bsh[buf][c * 16 + l15][8 * half];
      bop[c] = cat16(*(const v8bf*)bp, *(const v8bf*)(bp + 16));
    }
#pragma unroll
    for (int c = 0; c < 4; ++c)
      acc[c] = __builtin_amdgcn_wmma_f32_16x16x32_bf16(
          false, aop, false, bop[c], (short)0, acc[c], false, false);
    aop = aop_n;

    // desired pipeline: A vmem reads -> all 8 B ds reads -> 4 WMMAs
    SCHED_GROUP(0x020, 2, 0);
    SCHED_GROUP(0x100, 8, 0);
    SCHED_GROUP(0x008, 4, 0);
#ifdef USE_TDM
    if (w == 0) __builtin_amdgcn_s_wait_tensorcnt(0);
#else
    *(v8bf*)(&bsh[buf ^ 1][fn][fk]) = bnext;
#endif
    __syncthreads();
  }

#pragma unroll
  for (int c = 0; c < 4; ++c) {
    const int n  = n0 + c * 16 + l15;
    const float bv = bias[n];
#pragma unroll
    for (int r = 0; r < 8; ++r) {
      const int m = m0 + r + 8 * half;         // C layout: M = r + 8*(lane>=16)
      const float val = (acc[c][r] + bv) * oscale;
      if (mode == 2) {
        out_f[(size_t)m * D_MODEL + n] = val;
      } else {
        const int b = m >> 11, l = m & (SEQ - 1);
        const int h = n >> 6,  d = n & 63;
        if (mode == 0)
          out_bf[(((size_t)(b * N_HEAD + h) * SEQ) + l) * D_HEAD + d] = (bf16)val;
        else
          out_bf[(((size_t)(b * N_HEAD + h) * D_HEAD) + d) * SEQ + l] = (bf16)val;
      }
    }
  }
}

// ------------------------- flash attention ----------------------------------
// qh (pre-scaled by 1/8), kh: [B,H,L,64] bf16 ; vt: [B,H,64,L] bf16
// ao: [B*L, 1024] bf16.  Block = 8 waves = 8 query tiles of the SAME (b,h);
// K/V tiles staged in double-buffered LDS via TDM, shared by the block.
__global__ __launch_bounds__(256, 2) void k_flash_attn(
    const bf16* __restrict__ qh, const bf16* __restrict__ kh,
    const bf16* __restrict__ vt, bf16* __restrict__ ao) {
  __shared__ bf16 kbuf[2][32][KPAD];   // [buf][key][d]
  __shared__ bf16 vbuf[2][64][BPAD];   // [buf][d][key]
  __shared__ bf16 pbuf[8][16][BPAD];   // per-wave P staging

  const int tid  = threadIdx.x;
  const int w    = tid >> 5;
  const int lane = tid & 31;
  const int half = lane >> 4;
  const int l15  = lane & 15;

  const int wg = blockIdx.x * 8 + w;          // 0..4095
  const int qt = wg & 127;                    // query tile (block-local consecutive)
  const int h  = (wg >> 7) & 15;
  const int b  = wg >> 11;

  const bf16* qbase = qh + (size_t)(b * N_HEAD + h) * SEQ * D_HEAD;
  const bf16* kbase = kh + (size_t)(b * N_HEAD + h) * SEQ * D_HEAD;
  const bf16* vbase = vt + (size_t)(b * N_HEAD + h) * D_HEAD * SEQ;

#ifdef USE_TDM
  const unsigned ldsK0 = (unsigned)(size_t)&kbuf[0][0][0];
  const unsigned ldsK1 = (unsigned)(size_t)&kbuf[1][0][0];
  const unsigned ldsV0 = (unsigned)(size_t)&vbuf[0][0][0];
  const unsigned ldsV1 = (unsigned)(size_t)&vbuf[1][0][0];
  const unsigned long long gK = (unsigned long long)(size_t)kbase;
  const unsigned long long gV = (unsigned long long)(size_t)vbase;
#else
  const int kf_key = tid >> 3;                // 0..31
  const int kf_d   = (tid & 7) * 8;           // 0..56
  const int vf_d   = tid >> 2;                // 0..63
  const int vf_k   = (tid & 3) * 8;           // 0,8,16,24
  const bf16* ksrc = kbase + (size_t)kf_key * D_HEAD + kf_d;
  const bf16* vsrc = vbase + (size_t)vf_d * SEQ + vf_k;
#endif

  // Q as WMMA A operand for both d k-steps (already scaled by 1/sqrt(64))
  const bf16* qrow = qbase + (size_t)(qt * 16 + l15) * D_HEAD;
  const v16bf qA0 = cat16(*(const v8bf*)(qrow + 8 * half),
                          *(const v8bf*)(qrow + 16 + 8 * half));
  const v16bf qA1 = cat16(*(const v8bf*)(qrow + 32 + 8 * half),
                          *(const v8bf*)(qrow + 48 + 8 * half));

  v8f acc[4] = {};
  float mrun[8], lrun[8];
#pragma unroll
  for (int r = 0; r < 8; ++r) { mrun[r] = -1e30f; lrun[r] = 0.f; }

#ifdef USE_TDM
  if (w == 0) {
    tdm_load_2d(ldsK0, gK, TDM_FLAGS_PAD32, 64, 32, 64, 32, D_HEAD);
    tdm_load_2d(ldsV0, gV, TDM_FLAGS_PAD16, 32, 64, 32, 64, SEQ);
    __builtin_amdgcn_s_wait_tensorcnt(0);
  }
  __syncthreads();
#else
  *(v8bf*)(&kbuf[0][kf_key][kf_d]) = *(const v8bf*)(ksrc);
  *(v8bf*)(&vbuf[0][vf_d][vf_k])   = *(const v8bf*)(vsrc);
  __syncthreads();
#endif

  int buf = 0;
  for (int j0 = 0; j0 < SEQ; j0 += 32, buf ^= 1) {
#ifdef USE_TDM
    if (w == 0 && j0 + 32 < SEQ) {
      tdm_load_2d(buf ? ldsK0 : ldsK1, gK + (unsigned)((j0 + 32) * D_HEAD * 2),
                  TDM_FLAGS_PAD32, 64, 32, 64, 32, D_HEAD);
      tdm_load_2d(buf ? ldsV0 : ldsV1, gV + (unsigned)((j0 + 32) * 2),
                  TDM_FLAGS_PAD16, 32, 64, 32, 64, SEQ);
    }
#else
    const int j0n = (j0 + 32 < SEQ) ? j0 + 32 : 0;
    v8bf knext = *(const v8bf*)(ksrc + (size_t)j0n * D_HEAD);
    v8bf vnext = *(const v8bf*)(vsrc + j0n);
#endif

    // --- scores: hoist all 4 K-tile operands, then 4 WMMAs ---
    v16bf kop[4];                      // [t*2 + dstep]
#pragma unroll
    for (int t = 0; t < 2; ++t) {
      const bf16* krow = &kbuf[buf][t * 16 + l15][0];
      kop[t * 2 + 0] = cat16(*(const v8bf*)(krow + 8 * half),
                             *(const v8bf*)(krow + 8 * half + 16));
      kop[t * 2 + 1] = cat16(*(const v8bf*)(krow + 32 + 8 * half),
                             *(const v8bf*)(krow + 32 + 8 * half + 16));
    }
    v8f s[2];
#pragma unroll
    for (int t = 0; t < 2; ++t) {
      v8f st = {};
      st = __builtin_amdgcn_wmma_f32_16x16x32_bf16(false, qA0, false, kop[t * 2],
                                                   (short)0, st, false, false);
      st = __builtin_amdgcn_wmma_f32_16x16x32_bf16(false, qA1, false, kop[t * 2 + 1],
                                                   (short)0, st, false, false);
      s[t] = st;
    }
    // pipeline: all 8 K ds reads, then the 4 score WMMAs
    SCHED_GROUP(0x100, 8, 0);
    SCHED_GROUP(0x008, 4, 0);

    // --- online softmax (rows split across 16-lane halves per C layout) ---
    float corr[8];
    bf16* pw = &pbuf[w][0][0];
#pragma unroll
    for (int r = 0; r < 8; ++r) {
      float v0 = s[0][r];
      float v1 = s[1][r];
      float t = fmaxf(v0, v1);
      t = fmaxf(t, __shfl_xor(t, 1, 32));
      t = fmaxf(t, __shfl_xor(t, 2, 32));
      t = fmaxf(t, __shfl_xor(t, 4, 32));
      t = fmaxf(t, __shfl_xor(t, 8, 32));
      const float mnew = fmaxf(mrun[r], t);
      corr[r] = __expf(mrun[r] - mnew);
      mrun[r] = mnew;
      const float p0 = __expf(v0 - mnew);
      const float p1 = __expf(v1 - mnew);
      float rs = p0 + p1;
      rs += __shfl_xor(rs, 1, 32);
      rs += __shfl_xor(rs, 2, 32);
      rs += __shfl_xor(rs, 4, 32);
      rs += __shfl_xor(rs, 8, 32);
      lrun[r] = lrun[r] * corr[r] + rs;
      // stage P (16x32) to LDS: row m = r + 8*half, col = key-in-step
      pw[(r + 8 * half) * BPAD + l15]      = (bf16)p0;
      pw[(r + 8 * half) * BPAD + 16 + l15] = (bf16)p1;
    }
#pragma unroll
    for (int c = 0; c < 4; ++c)
#pragma unroll
      for (int r = 0; r < 8; ++r) acc[c][r] *= corr[r];

    // reload P in A-operand layout (same wave: DS ops are in-order)
    const bf16* pr = pw + l15 * BPAD;
    const v16bf pA = cat16(*(const v8bf*)(pr + 8 * half),
                           *(const v8bf*)(pr + 16 + 8 * half));

    // --- O += P * V : hoist all 4 V operands, then 4 WMMAs ---
    v16bf vop[4];
#pragma unroll
    for (int c = 0; c < 4; ++c) {
      const bf16* vrow = &vbuf[buf][c * 16 + l15][8 * half];
      vop[c] = cat16(*(const v8bf*)vrow, *(const v8bf*)(vrow + 16));
    }
#pragma unroll
    for (int c = 0; c < 4; ++c)
      acc[c] = __builtin_amdgcn_wmma_f32_16x16x32_bf16(
          false, pA, false, vop[c], (short)0, acc[c], false, false);
    // pipeline: P ds writes, then pA+V ds reads, then the 4 PV WMMAs
    SCHED_GROUP(0x200, 16, 0);
    SCHED_GROUP(0x100, 10, 0);
    SCHED_GROUP(0x008, 4, 0);

#ifdef USE_TDM
    if (w == 0) __builtin_amdgcn_s_wait_tensorcnt(0);
#else
    *(v8bf*)(&kbuf[buf ^ 1][kf_key][kf_d]) = knext;
    *(v8bf*)(&vbuf[buf ^ 1][vf_d][vf_k])   = vnext;
#endif
    __syncthreads();
  }

  // --- normalize and scatter to ao[B*L, 1024] ---
#pragma unroll
  for (int r = 0; r < 8; ++r) lrun[r] = 1.f / lrun[r];
#pragma unroll
  for (int c = 0; c < 4; ++c) {
    const int d = h * 64 + c * 16 + l15;
#pragma unroll
    for (int r = 0; r < 8; ++r) {
      const int row = b * SEQ + qt * 16 + r + 8 * half;
      ao[(size_t)row * D_MODEL + d] = (bf16)(acc[c][r] * lrun[r]);
    }
  }
}

// ---------------------------------------------------------------------------
extern "C" void kernel_launch(void* const* d_in, const int* in_sizes, int n_in,
                              void* d_out, int out_size, void* d_ws, size_t ws_size,
                              hipStream_t stream) {
  const float* q   = (const float*)d_in[0];
  const float* k   = (const float*)d_in[1];
  const float* v   = (const float*)d_in[2];
  const float* w_q = (const float*)d_in[3];
  const float* b_q = (const float*)d_in[4];
  const float* w_k = (const float*)d_in[5];
  const float* b_k = (const float*)d_in[6];
  const float* w_v = (const float*)d_in[7];
  const float* b_v = (const float*)d_in[8];
  const float* w_o = (const float*)d_in[9];
  const float* b_o = (const float*)d_in[10];
  float* out = (float*)d_out;

  const size_t ACT = (size_t)M_TOT * D_MODEL;           // 4M elements
  const size_t W   = (size_t)D_MODEL * D_MODEL;         // 1M elements
  char* ws = (char*)d_ws;
  bf16* qb  = (bf16*)(ws);                              // 8 MB each activation
  bf16* kb  = (bf16*)(ws + ACT * 2);
  bf16* vb  = (bf16*)(ws + ACT * 4);
  bf16* wqT = (bf16*)(ws + ACT * 6);                    // 2 MB each weight
  bf16* wkT = (bf16*)(ws + ACT * 6 + W * 2);
  bf16* wvT = (bf16*)(ws + ACT * 6 + W * 4);
  bf16* woT = (bf16*)(ws + ACT * 6 + W * 6);
  bf16* qhh = (bf16*)(ws + ACT * 6 + W * 8);            // [B,H,L,64], pre-scaled
  bf16* khh = (bf16*)(ws + ACT * 8 + W * 8);
  bf16* vtt = (bf16*)(ws + ACT * 10 + W * 8);           // [B,H,64,L]
  bf16* aob = (bf16*)(ws + ACT * 12 + W * 8);           // attention output bf16
  // requires 64 MB of d_ws

  // 1) convert activations
  k_cvt_bf16<<<(int)(ACT / (256 * 4)), 256, 0, stream>>>(q, qb, (int)ACT);
  k_cvt_bf16<<<(int)(ACT / (256 * 4)), 256, 0, stream>>>(k, kb, (int)ACT);
  k_cvt_bf16<<<(int)(ACT / (256 * 4)), 256, 0, stream>>>(v, vb, (int)ACT);
  // 2) transpose+convert weights
  k_wT_bf16<<<(int)(W / 256), 256, 0, stream>>>(w_q, wqT);
  k_wT_bf16<<<(int)(W / 256), 256, 0, stream>>>(w_k, wkT);
  k_wT_bf16<<<(int)(W / 256), 256, 0, stream>>>(w_v, wvT);
  k_wT_bf16<<<(int)(W / 256), 256, 0, stream>>>(w_o, woT);

  dim3 ggrid(M_TOT / 128, D_MODEL / 64);
  // 3) QKV projections (Q gets the 1/sqrt(d_head) fold)
  k_gemm_bf16<<<ggrid, 256, 0, stream>>>(qb, wqT, b_q, qhh, nullptr, 0, 0.125f);
  k_gemm_bf16<<<ggrid, 256, 0, stream>>>(kb, wkT, b_k, khh, nullptr, 0, 1.0f);
  k_gemm_bf16<<<ggrid, 256, 0, stream>>>(vb, wvT, b_v, vtt, nullptr, 1, 1.0f);
  // 4) attention: 4096 wave-tiles / 8 waves per block
  k_flash_attn<<<512, 256, 0, stream>>>(qhh, khh, vtt, aob);
  // 5) output projection (f32 + bias into d_out)
  k_gemm_bf16<<<ggrid, 256, 0, stream>>>(aob, woT, b_o, nullptr, out, 2, 1.0f);
}
